// LeakySNN_53815940219125
// MI455X (gfx1250) — matrix operand order repaired and checked by reading, not compiled, not run on hardware
//
#include <hip/hip_runtime.h>

// ---- problem constants (from reference) ----
#define NUM_STEPS 25
constexpr int B_ = 32768, NI = 128, NH = 256, NO = 32;

typedef __attribute__((ext_vector_type(16))) __bf16 v16bf;
typedef __attribute__((ext_vector_type(8)))  __bf16 v8bf;
typedef __attribute__((ext_vector_type(8)))  float  v8f;

__device__ __forceinline__ unsigned short f2bf_u(float f) {
  unsigned u = __builtin_bit_cast(unsigned, f);
  u += 0x7FFFu + ((u >> 16) & 1u);            // round-to-nearest-even
  return (unsigned short)(u >> 16);
}
__device__ __forceinline__ __bf16 f2bf(float f) {
  unsigned short s = f2bf_u(f);
  return __builtin_bit_cast(__bf16, s);
}

// Fused LIF scan.  Reference per step: reset=(mem_in>th); mem=bc*mem+spk-reset*th;
// spk=(mem>th).  Since reset at step t equals spk from step t-1 (same compare of the
// same mem), for steps>=2:  mem' = bc*mem + spk*(1-th).  Carry w = spk*(1-th)
// directly via the select, so each step is fma+cmp+cndmask (3 VALU).
__device__ __forceinline__ float lif_scan(float cur1_plus_badj, float bc, float th,
                                          float c1 /* = 1-th */) {
  float mem = cur1_plus_badj;                 // mem after step 1 (bias & (0>th)*th folded)
  float w = mem > th ? c1 : 0.f;              // spk1 * (1-th)
#pragma unroll
  for (int t = 1; t < NUM_STEPS; ++t) {       // steps 2..25
    mem = __builtin_fmaf(bc, mem, w);
    w = mem > th ? c1 : 0.f;
  }
  return mem > th ? 1.f : 0.f;                // spk_25
}

__global__ __launch_bounds__(256)
void snn_fused_kernel(const float* __restrict__ x,  const float* __restrict__ W1,
                      const float* __restrict__ b1, const float* __restrict__ W2,
                      const float* __restrict__ b2, const float* __restrict__ beta_p,
                      const float* __restrict__ th_p, float* __restrict__ out) {
  __shared__ unsigned short sW1[NH * NI];      // 64 KB: W1 as bf16, row-major [n][k]
  __shared__ unsigned short sW2[NO * NH];      // 16 KB: W2 as bf16, row-major [n][k]
  __shared__ float sB1[NH];
  __shared__ float sB2[NO];
  __shared__ unsigned short sSpk[8][16 * 32];  // 8 KB: per-wave C->A relayout scratch

  const int tid  = threadIdx.x;
  const int wave = tid >> 5;
  const int lane = tid & 31;
  const int half = lane >> 4;    // which 16-lane half of the wave
  const int col  = lane & 15;    // N (or M for A-frag) index within tile
  const int rowBase = blockIdx.x * 128 + wave * 16;

  // prefetch this wave's x tile while weights stage (global_prefetch_b8)
  __builtin_prefetch(&x[(rowBase + col) * NI], 0, 0);

  // ---- stage W1/W2 (f32 -> bf16) into LDS, cooperatively ----
  for (int i = tid; i < NH * NI / 4; i += 256) {
    float4 f = reinterpret_cast<const float4*>(W1)[i];
    uint2 p;
    p.x = (unsigned)f2bf_u(f.x) | ((unsigned)f2bf_u(f.y) << 16);
    p.y = (unsigned)f2bf_u(f.z) | ((unsigned)f2bf_u(f.w) << 16);
    reinterpret_cast<uint2*>(sW1)[i] = p;
  }
  for (int i = tid; i < NO * NH / 4; i += 256) {
    float4 f = reinterpret_cast<const float4*>(W2)[i];
    uint2 p;
    p.x = (unsigned)f2bf_u(f.x) | ((unsigned)f2bf_u(f.y) << 16);
    p.y = (unsigned)f2bf_u(f.z) | ((unsigned)f2bf_u(f.w) << 16);
    reinterpret_cast<uint2*>(sW2)[i] = p;
  }
  if (tid < NH) sB1[tid] = b1[tid];
  if (tid < NO) sB2[tid] = b2[tid];
  __syncthreads();

  const float th = th_p[0];
  float bc = beta_p[0];
  bc = bc < 0.f ? 0.f : (bc > 1.f ? 1.f : bc);   // clip(beta, 0, 1)
  const float c1   = 1.f - th;                    // spike contribution (1 - th)
  const float r0th = (0.f > th) ? th : 0.f;       // step-1 reset correction

  // ---- load A1 fragments: 16 rows of x, K=128 as 4 chunks of 32 (bf16) ----
  // A fragment (16x32), this code's contiguous k order: lanes 0-15 hold M=0..15; elem e -> K=kb+half*16+e
  const int rowOff = (rowBase + col) * NI;
  v16bf a1[4];
#pragma unroll
  for (int kc = 0; kc < 4; ++kc) {
    const int kb = kc * 32;
    const float4* p0 = reinterpret_cast<const float4*>(&x[rowOff + kb + half * 16]);
    const float4* p1 = reinterpret_cast<const float4*>(&x[rowOff + kb + half * 16 + 8]);
    float4 f0 = p0[0], f1 = p0[1], f2 = p1[0], f3 = p1[1];
    v16bf a;
    a[0] = f2bf(f0.x); a[1]  = f2bf(f0.y); a[2]  = f2bf(f0.z); a[3]  = f2bf(f0.w);
    a[4] = f2bf(f1.x); a[5]  = f2bf(f1.y); a[6]  = f2bf(f1.z); a[7]  = f2bf(f1.w);
    a[8] = f2bf(f2.x); a[9]  = f2bf(f2.y); a[10] = f2bf(f2.z); a[11] = f2bf(f2.w);
    a[12]= f2bf(f3.x); a[13] = f2bf(f3.y); a[14] = f2bf(f3.z); a[15] = f2bf(f3.w);
    a1[kc] = a;
  }

  v8f out0 = {}; v8f out1 = {};            // final 16x32 output accumulators
  unsigned short* scr = &sSpk[wave][0];

  // ---- hidden dim in chunks of 32 (two 16-wide tiles) ----
  for (int hc = 0; hc < 8; ++hc) {
    const int hh = hc * 32;
    v8f acc0 = {}, acc1 = {};
    // GEMM1: cur1 tile = x(16x128) @ W1^T chunk.  B[k][n] = W1[n][k]:
    // lane gives n = base+col; elem e -> K = kc*32 + half*16 + e (contiguous in LDS).
#pragma unroll
    for (int kc = 0; kc < 4; ++kc) {
      const int kbase = kc * 32 + half * 16;
      const v16bf bA = *reinterpret_cast<const v16bf*>(&sW1[(hh + col) * NI + kbase]);
      const v16bf bB = *reinterpret_cast<const v16bf*>(&sW1[(hh + 16 + col) * NI + kbase]);
      acc0 = __builtin_amdgcn_wmma_f32_16x16x32_bf16(false, a1[kc], false, bA,
                                                     (short)0, acc0, false, false);
      acc1 = __builtin_amdgcn_wmma_f32_16x16x32_bf16(false, a1[kc], false, bB,
                                                     (short)0, acc1, false, false);
    }
    // fold bias and the step-1 reset correction into one scalar per column
    const float badj0 = sB1[hh + col]      - r0th;
    const float badj1 = sB1[hh + 16 + col] - r0th;

    // ---- 25-step LIF recurrence (3 VALU/step/elem), 16 independent chains ----
#pragma unroll
    for (int r = 0; r < 8; ++r) {
      acc0[r] = lif_scan(acc0[r] + badj0, bc, th, c1);
      acc1[r] = lif_scan(acc1[r] + badj1, bc, th, c1);
    }

    // ---- relayout spk tile C-layout -> A-layout via per-wave LDS scratch ----
    // C-layout: elem r is row (r + 8*half), column col (+16 for second tile).
#pragma unroll
    for (int r = 0; r < 8; ++r) {
      const int rr = r + half * 8;
      scr[rr * 32 + col]      = f2bf_u(acc0[r]);   // 0.0 / 1.0 are exact in bf16
      scr[rr * 32 + 16 + col] = f2bf_u(acc1[r]);
    }
    __builtin_amdgcn_wave_barrier();
    asm volatile("s_wait_dscnt 0" ::: "memory");   // wave-synchronous handoff

    v16bf a2;
    {
      const v8bf lo = *reinterpret_cast<const v8bf*>(&scr[col * 32 + half * 16]);
      const v8bf hi = *reinterpret_cast<const v8bf*>(&scr[col * 32 + half * 16 + 8]);
#pragma unroll
      for (int e = 0; e < 8; ++e) { a2[e] = lo[e]; a2[8 + e] = hi[e]; }
    }
    __builtin_amdgcn_wave_barrier();               // keep reads before next chunk's writes

    // GEMM2 accumulate: out += spk_chunk(16x32) @ W2^T chunk.  B[k][n] = W2[n][hh+k].
    const int kb2 = hh + half * 16;
    const v16bf w2a = *reinterpret_cast<const v16bf*>(&sW2[col * NH + kb2]);
    const v16bf w2b = *reinterpret_cast<const v16bf*>(&sW2[(16 + col) * NH + kb2]);
    out0 = __builtin_amdgcn_wmma_f32_16x16x32_bf16(false, a2, false, w2a,
                                                   (short)0, out0, false, false);
    out1 = __builtin_amdgcn_wmma_f32_16x16x32_bf16(false, a2, false, w2b,
                                                   (short)0, out1, false, false);
  }

  // ---- epilogue: bias + store (row r+8*half, col / col+16) ----
  const float ob0 = sB2[col];
  const float ob1 = sB2[16 + col];
#pragma unroll
  for (int r = 0; r < 8; ++r) {
    const int row = rowBase + r + half * 8;
    out[row * NO + col]      = out0[r] + ob0;
    out[row * NO + 16 + col] = out1[r] + ob1;
  }
}

extern "C" void kernel_launch(void* const* d_in, const int* in_sizes, int n_in,
                              void* d_out, int out_size, void* d_ws, size_t ws_size,
                              hipStream_t stream) {
  (void)in_sizes; (void)n_in; (void)d_ws; (void)ws_size; (void)out_size;
  const float* x    = (const float*)d_in[0];
  const float* W1   = (const float*)d_in[1];
  const float* b1   = (const float*)d_in[2];
  const float* W2   = (const float*)d_in[3];
  const float* b2   = (const float*)d_in[4];
  const float* beta = (const float*)d_in[5];
  const float* th   = (const float*)d_in[6];
  float* out = (float*)d_out;

  dim3 grid(B_ / 128);   // 8 waves/block * 16 rows/wave = 128 rows/block
  dim3 block(256);
  hipLaunchKernelGGL(snn_fused_kernel, grid, block, 0, stream,
                     x, W1, b1, W2, b2, beta, th, out);
}
